// Head_36893769073054
// MI455X (gfx1250) — matrix operand order repaired and checked by reading, not compiled, not run on hardware
//
#include <hip/hip_runtime.h>
#include <hip/hip_bf16.h>
#include <math.h>
#include <stdint.h>

// ---------------------------------------------------------------------------
// MI455X (gfx1250) fused DAG-attention, bf16 WMMA everywhere, f32 epilogues.
//   K,Q,V = swish(X W^T + b)              (K stored [B,T,H]; Q,V stored [B,H,T])
//   Q' = dagT @ Q ; V' = dagT @ V         (rewrite dagT@(Q K^T) = (dagT@Q) K^T)
//   S  = mask(Q' K^T / sqrt(H)) in LDS ; O = softmax(S) V + V'
// LDS fragment feeds are contiguous 16B chunks (2x ds_load_b128 per fragment).
// Pure-copy stagings use GLOBAL_LOAD_ASYNC_TO_LDS_B128 with DOUBLE BUFFERING:
// issue next tile's async group, then s_wait_asynccnt <G> (in-order completion
// proves the previous group is resident) so the DMA overlaps the WMMA pipe.
// ---------------------------------------------------------------------------

typedef __bf16 bf16;
typedef __attribute__((ext_vector_type(16))) __bf16 v16bf;
typedef __attribute__((ext_vector_type(8)))  __bf16 v8bf;
typedef __attribute__((ext_vector_type(4)))  __bf16 v4bf;
typedef __attribute__((ext_vector_type(8)))  float  v8f;

#define NEG_INF (-__builtin_inff())

constexpr int Bc = 64, Tc = 512, Dc = 512, Hc = 128;

// ---- async global->LDS 16B copy (ISA 15.18.3 op 98, ASYNCcnt-tracked) ------
__device__ __forceinline__ void async_copy16(void* lds_dst, const void* gsrc) {
  unsigned           l = (unsigned)(uintptr_t)lds_dst;   // addr[31:0] = LDS offset
  unsigned long long g = (unsigned long long)(uintptr_t)gsrc;
  asm volatile("global_load_async_to_lds_b128 %0, %1, off"
               :: "v"(l), "v"(g) : "memory");
}
// Wait until ASYNCcnt <= n (n = ops issued for the in-flight NEXT tile).
#define ASYNC_WAIT(n) asm volatile("s_wait_asynccnt " #n ::: "memory")

// A fragment: 16x32 bf16 tile of row-major A[.][lda] at (m0,k0).
__device__ __forceinline__ v16bf load_a_frag(const bf16* A, int lda, int m0, int k0, int lane) {
  int m  = m0 + (lane & 15);
  int kh = ((lane >> 4) & 1) * 8;
  const bf16* p = A + m * lda + k0 + kh;
  v8bf lo = *(const v8bf*)(p);
  v8bf hi = *(const v8bf*)(p + 16);
  return __builtin_shufflevector(lo, hi, 0,1,2,3,4,5,6,7,8,9,10,11,12,13,14,15);
}
// B fragment from LDS stored TRANSPOSED: Bt[N][ldb] (columns = Kdim).
__device__ __forceinline__ v16bf load_b_frag(const bf16* Bt, int ldb, int k0, int n0, int lane) {
  int n  = n0 + (lane & 15);
  int kh = k0 + ((lane >> 4) & 1) * 8;
  const bf16* p = Bt + n * ldb + kh;
  v8bf lo = *(const v8bf*)(p);
  v8bf hi = *(const v8bf*)(p + 16);
  return __builtin_shufflevector(lo, hi, 0,1,2,3,4,5,6,7,8,9,10,11,12,13,14,15);
}
__device__ __forceinline__ v8f wmma_bf16(v16bf a, v16bf b, v8f c) {
  return __builtin_amdgcn_wmma_f32_16x16x32_bf16(false, a, false, b, (short)0, c, false, false);
}
__device__ __forceinline__ v4bf cvt4(float4 x) {
  v4bf y; y[0] = (bf16)x.x; y[1] = (bf16)x.y; y[2] = (bf16)x.z; y[3] = (bf16)x.w;
  return y;
}
__device__ __forceinline__ float swishf(float c) {
  return c * __builtin_amdgcn_rcpf(1.f + __expf(-c));
}

// ---- dag -> dagT (bf16, exact for 0/1) -------------------------------------
__global__ __launch_bounds__(256) void k_dagT(const float* __restrict__ dag,
                                              bf16* __restrict__ dagTb) {
  int f = blockIdx.x * 256 + threadIdx.x;   // f = i*T + j
  int i = f >> 9, j = f & 511;
  dagTb[f] = (bf16)dag[j * Tc + i];
}

// ---- K/Q/V projection + swish ----------------------------------------------
// grid (BT/64, 1, 3); block 256 (8 waves = 4M x 2N; wave tile 16x64).
// f32 sources need in-flight cvt, so this stays on the VGPR staging path.
__global__ __launch_bounds__(256) void k_qkv(
    const float* __restrict__ X,
    const float* __restrict__ Wk, const float* __restrict__ bk,
    const float* __restrict__ Wq, const float* __restrict__ bq,
    const float* __restrict__ Wv, const float* __restrict__ bv,
    bf16* __restrict__ Kb, bf16* __restrict__ QbT, bf16* __restrict__ VbT) {
  const float* W; const float* bias; bf16* out;
  switch (blockIdx.z) {
    case 0:  W = Wk; bias = bk; out = Kb;  break;
    case 1:  W = Wq; bias = bq; out = QbT; break;
    default: W = Wv; bias = bv; out = VbT; break;
  }
  __shared__ bf16 As[64 * 40];     // X tile 64x32 row-major
  __shared__ bf16 Bt[128 * 40];    // W tile as [N=128][K=32]
  __shared__ bf16 OS[128 * 72];    // transposed output staging [h][m]

  const int tid = threadIdx.x, lane = tid & 31, wave = tid >> 5;
  const int row0 = blockIdx.x * 64;
  const int m0 = (wave & 3) * 16, nbase = (wave >> 2) * 64;

  v8f acc[4] = {};
  for (int k0 = 0; k0 < Dc; k0 += 32) {
#pragma unroll
    for (int i = 0; i < 2; ++i) {
      int q = tid + 256 * i, r = q >> 3, c = (q & 7) * 4;
      float4 x = *(const float4*)&X[(size_t)(row0 + r) * Dc + k0 + c];
      *(v4bf*)&As[r * 40 + c] = cvt4(x);
    }
#pragma unroll
    for (int i = 0; i < 4; ++i) {
      int q = tid + 256 * i, n = q >> 3, c = (q & 7) * 4;
      float4 w = *(const float4*)&W[n * Dc + k0 + c];
      *(v4bf*)&Bt[n * 40 + c] = cvt4(w);
    }
    __syncthreads();
    v16bf a = load_a_frag(As, 40, m0, 0, lane);
#pragma unroll
    for (int j = 0; j < 4; ++j) {
      v16bf b = load_b_frag(Bt, 40, 0, nbase + 16 * j, lane);
      acc[j] = wmma_bf16(a, b, acc[j]);
    }
    __syncthreads();
  }
  if (blockIdx.z == 0) {                           // K: direct [B,T,H] stores
#pragma unroll
    for (int j = 0; j < 4; ++j)
#pragma unroll
      for (int r = 0; r < 8; ++r) {
        int m = m0 + r + 8 * (lane >> 4);
        int n = nbase + 16 * j + (lane & 15);
        out[(size_t)(row0 + m) * Hc + n] = (bf16)swishf(acc[j][r] + bias[n]);
      }
  } else {                                         // Q/V: transpose via LDS
#pragma unroll
    for (int j = 0; j < 4; ++j)
#pragma unroll
      for (int r = 0; r < 8; ++r) {
        int m = m0 + r + 8 * (lane >> 4);
        int n = nbase + 16 * j + (lane & 15);
        OS[n * 72 + m] = (bf16)swishf(acc[j][r] + bias[n]);
      }
    __syncthreads();
    int bb = row0 >> 9, t0 = row0 & 511;
#pragma unroll
    for (int i = 0; i < 4; ++i) {
      int idx = tid + 256 * i, h = idx >> 3, m8 = (idx & 7) * 8;
      *(v8bf*)&out[((size_t)bb * Hc + h) * Tc + t0 + m8] = *(const v8bf*)&OS[h * 72 + m8];
    }
  }
}

// ---- Q' = dagT @ Q (bf16 [B,T,H]), V' = dagT @ V (f32 [B,T,H]) -------------
// grid (T/64, B, 2); block 256 (4M x 2N waves; wave tile 16x64).
// Double-buffered async staging: 3 async ops/wave/tile -> s_wait_asynccnt 3.
__global__ __launch_bounds__(256) void k_dagmm(
    const bf16* __restrict__ dagTb, const bf16* __restrict__ QbT,
    const bf16* __restrict__ VbT, bf16* __restrict__ Qpb, float* __restrict__ Vp) {
  __shared__ bf16 As[2][64 * 40];
  __shared__ bf16 Bt[2][128 * 40];
  const int tid = threadIdx.x, lane = tid & 31, wave = tid >> 5;
  const int b = blockIdx.y, i0 = blockIdx.x * 64, z = blockIdx.z;
  const bf16* srcT = z ? VbT : QbT;
  const int m0 = (wave & 3) * 16, nbase = (wave >> 2) * 64;

  auto issue = [&](int buf, int k0) {              // 3 async ops per wave
    int r = tid >> 2, c8 = (tid & 3) * 8;
    async_copy16(&As[buf][r * 40 + c8], &dagTb[(size_t)(i0 + r) * Tc + k0 + c8]);
#pragma unroll
    for (int i = 0; i < 2; ++i) {
      int idx = tid + 256 * i, n = idx >> 2, cb = (idx & 3) * 8;
      async_copy16(&Bt[buf][n * 40 + cb], &srcT[((size_t)b * Hc + n) * Tc + k0 + cb]);
    }
  };

  v8f acc[4] = {};
  issue(0, 0);
  int p = 0;
  for (int k0 = 0; k0 < Tc; k0 += 32) {
    if (k0 + 32 < Tc) { issue(p ^ 1, k0 + 32); ASYNC_WAIT(3); }
    else              { ASYNC_WAIT(0); }
    __syncthreads();                               // tile p resident everywhere
    v16bf a = load_a_frag(As[p], 40, m0, 0, lane);
#pragma unroll
    for (int j = 0; j < 4; ++j) {
      v16bf bb = load_b_frag(Bt[p], 40, 0, nbase + 16 * j, lane);
      acc[j] = wmma_bf16(a, bb, acc[j]);
    }
    __syncthreads();                               // tile p consumed
    p ^= 1;
  }
#pragma unroll
  for (int j = 0; j < 4; ++j)
#pragma unroll
    for (int r = 0; r < 8; ++r) {
      int m = m0 + r + 8 * (lane >> 4);
      int n = nbase + 16 * j + (lane & 15);
      size_t gi = ((size_t)b * Tc + i0 + m) * Hc + n;
      if (z == 0) Qpb[gi] = (bf16)acc[j][r];
      else        Vp[gi]  = acc[j][r];
    }
}

// ---- fused attention: S in LDS, masked softmax, O = P V / sum + V' ---------
// grid (T/32, B); block 256; waves as 2M x 4N. Double-buffered async K/V tiles.
__global__ __launch_bounds__(256) void k_attn(
    const bf16* __restrict__ Qpb, const bf16* __restrict__ Kb,
    const bf16* __restrict__ VbT, const bf16* __restrict__ dagTb,
    const float* __restrict__ Vp, float* __restrict__ O) {
  __shared__ bf16  Aq[32 * 136];      // Q' tile 32x128 row-major
  __shared__ bf16  Bsh[2][64 * 136];  // phase1: [kcol=64][h=128]; phase3: [h=128][t=32]
  __shared__ float Ssh[32 * 520];     // score tile (f32)
  __shared__ bf16  Psh[32 * 520];     // unnormalized probs (bf16)
  __shared__ float rowscale[32];

  const int tid = threadIdx.x, lane = tid & 31, wave = tid >> 5;
  const int b = blockIdx.y, i0 = blockIdx.x * 32;
  const float scale = 0.08838834764831845f;   // 1/sqrt(128)
  const int m0 = (wave & 1) * 16, waveN = wave >> 1;

  auto issueK = [&](int buf, int k0) {             // 4 async ops per wave
#pragma unroll
    for (int i = 0; i < 4; ++i) {
      int idx = tid + 256 * i, kk = idx >> 4, c8 = (idx & 15) * 8;
      async_copy16(&Bsh[buf][kk * 136 + c8], &Kb[((size_t)b * Tc + k0 + kk) * Hc + c8]);
    }
  };
  auto issueV = [&](int buf, int kb) {             // 2 async ops per wave
#pragma unroll
    for (int i = 0; i < 2; ++i) {
      int idx = tid + 256 * i, h = idx >> 2, c8 = (idx & 3) * 8;
      async_copy16(&Bsh[buf][h * 40 + c8], &VbT[((size_t)b * Hc + h) * Tc + kb + c8]);
    }
  };

  // prologue: Q' tile (2 ops) + first K tile (4 ops)
#pragma unroll
  for (int i = 0; i < 2; ++i) {
    int idx = tid + 256 * i, r = idx >> 4, c8 = (idx & 15) * 8;
    async_copy16(&Aq[r * 136 + c8], &Qpb[((size_t)b * Tc + i0 + r) * Hc + c8]);
  }
  issueK(0, 0);

  // ---- phase 1: S = mask(Q' K^T * scale) into LDS ----
  int p = 0;
  for (int k0 = 0; k0 < Tc; k0 += 64) {
    if (k0 + 64 < Tc) { issueK(p ^ 1, k0 + 64); ASYNC_WAIT(4); }  // Aq+tile p done
    else              { ASYNC_WAIT(0); }
    __syncthreads();
    v8f acc = {};
#pragma unroll
    for (int kk = 0; kk < 4; ++kk) {          // Kdim = H = 128
      v16bf a  = load_a_frag(Aq, 136, m0, kk * 32, lane);
      v16bf bb = load_b_frag(Bsh[p], 136, kk * 32, waveN * 16, lane);
      acc = wmma_bf16(a, bb, acc);
    }
#pragma unroll
    for (int r = 0; r < 8; ++r) {
      int m = m0 + r + 8 * (lane >> 4);
      int kloc = k0 + waveN * 16 + (lane & 15);
      float s = acc[r] * scale;
      float d = (float)dagTb[(size_t)(i0 + m) * Tc + kloc];
      Ssh[m * 520 + kloc] = (d != 0.f && s != 0.f) ? s : NEG_INF;
    }
    __syncthreads();
    p ^= 1;
  }

  // ---- phase 2: row softmax; per-lane contiguous 16-elem chunk ----
#pragma unroll
  for (int rr = 0; rr < 4; ++rr) {
    int r = wave * 4 + rr;
    const float4* srow = (const float4*)&Ssh[r * 520 + lane * 16];
    float4 s4[4];
    float mx = NEG_INF;
#pragma unroll
    for (int q = 0; q < 4; ++q) {
      s4[q] = srow[q];
      mx = fmaxf(mx, fmaxf(fmaxf(s4[q].x, s4[q].y), fmaxf(s4[q].z, s4[q].w)));
    }
#pragma unroll
    for (int off = 16; off > 0; off >>= 1) mx = fmaxf(mx, __shfl_xor(mx, off, 32));
    bool dead = (mx == NEG_INF);
    float sum = 0.f;
    v8bf p0, p1;
#pragma unroll
    for (int q = 0; q < 4; ++q) {
      float e0 = dead ? 0.f : __expf(s4[q].x - mx);
      float e1 = dead ? 0.f : __expf(s4[q].y - mx);
      float e2 = dead ? 0.f : __expf(s4[q].z - mx);
      float e3 = dead ? 0.f : __expf(s4[q].w - mx);
      sum += (e0 + e1) + (e2 + e3);
      v8bf& pd = (q < 2) ? p0 : p1;
      int o = (q & 1) * 4;
      pd[o] = (bf16)e0; pd[o+1] = (bf16)e1; pd[o+2] = (bf16)e2; pd[o+3] = (bf16)e3;
    }
    *(v8bf*)&Psh[r * 520 + lane * 16]     = p0;
    *(v8bf*)&Psh[r * 520 + lane * 16 + 8] = p1;
#pragma unroll
    for (int off = 16; off > 0; off >>= 1) sum += __shfl_xor(sum, off, 32);
    if (lane == 0) rowscale[r] = dead ? 0.f : 1.f / sum;
  }
  __syncthreads();

  // ---- phase 3: O = (P V) * rowscale + V' ----
  const int n0 = (wave >> 1) * 32;
  v8f acc2[2] = {};
  issueV(0, 0);
  int pv = 0;
  for (int kb = 0; kb < Tc; kb += 32) {
    if (kb + 32 < Tc) { issueV(pv ^ 1, kb + 32); ASYNC_WAIT(2); }
    else              { ASYNC_WAIT(0); }
    __syncthreads();
    v16bf a = load_a_frag(Psh, 520, m0, kb, lane);
#pragma unroll
    for (int j = 0; j < 2; ++j) {
      v16bf bb = load_b_frag(Bsh[pv], 40, 0, n0 + 16 * j, lane);
      acc2[j] = wmma_bf16(a, bb, acc2[j]);
    }
    __syncthreads();
    pv ^= 1;
  }
#pragma unroll
  for (int j = 0; j < 2; ++j)
#pragma unroll
    for (int r = 0; r < 8; ++r) {
      int m = m0 + r + 8 * (lane >> 4);
      int n = n0 + 16 * j + (lane & 15);
      size_t gi = ((size_t)b * Tc + i0 + m) * Hc + n;
      O[gi] = acc2[j][r] * rowscale[m] + Vp[gi];
    }
}

// ---------------------------------------------------------------------------
extern "C" void kernel_launch(void* const* d_in, const int* in_sizes, int n_in,
                              void* d_out, int out_size, void* d_ws, size_t ws_size,
                              hipStream_t stream) {
  (void)in_sizes; (void)n_in; (void)out_size; (void)ws_size;
  const float* X   = (const float*)d_in[0];
  const float* dag = (const float*)d_in[1];
  const float* Wk  = (const float*)d_in[2];
  const float* bk  = (const float*)d_in[3];
  const float* Wq  = (const float*)d_in[4];
  const float* bq  = (const float*)d_in[5];
  const float* Wv  = (const float*)d_in[6];
  const float* bv  = (const float*)d_in[7];

  char* ws = (char*)d_ws;
  const size_t SZ_BF = (size_t)Bc * Tc * Hc * sizeof(bf16);   // 8 MiB
  const size_t SZ_F  = (size_t)Bc * Tc * Hc * sizeof(float);  // 16 MiB
  bf16*  Kb    = (bf16*)(ws);                      // [B,T,H]
  bf16*  QbT   = (bf16*)(ws + SZ_BF);              // [B,H,T]
  bf16*  VbT   = (bf16*)(ws + 2 * SZ_BF);          // [B,H,T]
  bf16*  Qpb   = (bf16*)(ws + 3 * SZ_BF);          // [B,T,H]
  float* Vp    = (float*)(ws + 4 * SZ_BF);         // [B,T,H] f32
  bf16*  dagTb = (bf16*)(ws + 4 * SZ_BF + SZ_F);   // [T,T]

  k_dagT<<<(Tc * Tc) / 256, 256, 0, stream>>>(dag, dagTb);
  k_qkv<<<dim3((Bc * Tc) / 64, 1, 3), 256, 0, stream>>>(X, Wk, bk, Wq, bq, Wv, bv,
                                                        Kb, QbT, VbT);
  k_dagmm<<<dim3(Tc / 64, Bc, 2), 256, 0, stream>>>(dagTb, QbT, VbT, Qpb, Vp);
  k_attn<<<dim3(Tc / 32, Bc), 256, 0, stream>>>(Qpb, Kb, VbT, dagTb, Vp, (float*)d_out);
}
